// TruePatchTSTClassification_18330920419391
// MI455X (gfx1250) — compile-verified, matrix operand order checked by reading, not run on hardware
//
#include <hip/hip_runtime.h>
#include <hip/hip_bf16.h>
#include <math.h>

// ---------------------------------------------------------------- constants
#define NB   128      // batch
#define SEQ  2048     // sequence length
#define NP   255      // patches  ((2048-16)/8 + 1)
#define DIM  512      // model dim
#define NH   8        // heads
#define DHD  64       // head dim
#define FFD  2048     // ffn dim
#define NL   4        // layers
#define N3D  1536     // 3*DIM
#define MTOK (NB*NP)  // 32640 flattened tokens (multiple of 64)

typedef __attribute__((ext_vector_type(16))) __bf16       bf16x16;
typedef __attribute__((ext_vector_type(8)))  float        f32x8;
typedef __attribute__((ext_vector_type(4)))  unsigned int u32x4;

struct Frag256 { u32x4 lo, hi; };  // 32B raw bits -> bf16x16

__device__ __forceinline__ unsigned short f2bf(float f) {
  unsigned u = __builtin_bit_cast(unsigned, f);
  unsigned r = u + 0x7FFFu + ((u >> 16) & 1u);   // round-to-nearest-even
  return (unsigned short)(r >> 16);
}

__device__ __forceinline__ bf16x16 load_frag(const unsigned short* p0,
                                             const unsigned short* p1) {
  Frag256 f;
  f.lo = *(const u32x4*)p0;   // 16B -> elems 0..7
  f.hi = *(const u32x4*)p1;   // 16B -> elems 8..15
  return __builtin_bit_cast(bf16x16, f);
}

__device__ __forceinline__ float gelu_exact(float x) {
  return 0.5f * x * (1.f + erff(x * 0.70710678118f));
}

// ---------------------------------------------------------------- to bf16
__global__ void to_bf16k(const float* __restrict__ s, unsigned short* __restrict__ d,
                         size_t n) {
  size_t i = (size_t)blockIdx.x * blockDim.x + threadIdx.x;
  if (i < n) d[i] = f2bf(s[i]);
}

// ------------------------------------------------- channel-3 normalization
__global__ __launch_bounds__(256)
void close_norm(const float* __restrict__ x, float* __restrict__ cl) {
  const int b = blockIdx.x, tid = threadIdx.x;
  __shared__ float rs[256], rq[256];
  float s = 0.f, q = 0.f;
  for (int i = tid; i < SEQ; i += 256) {
    float v = x[((size_t)b * SEQ + i) * 8 + 3];
    s += v; q += v * v;
  }
  rs[tid] = s; rq[tid] = q; __syncthreads();
  for (int off = 128; off > 0; off >>= 1) {
    if (tid < off) { rs[tid] += rs[tid + off]; rq[tid] += rq[tid + off]; }
    __syncthreads();
  }
  float mean = rs[0] / (float)SEQ;
  float var  = rq[0] / (float)SEQ - mean * mean;
  float inv  = 1.f / (sqrtf(fmaxf(var, 0.f)) + 1e-8f);
  for (int i = tid; i < SEQ; i += 256)
    cl[(size_t)b * SEQ + i] = (x[((size_t)b * SEQ + i) * 8 + 3] - mean) * inv;
}

// ------------------------------------------------ patch embed + sinus PE
__global__ __launch_bounds__(256)
void patch_embed(const float* __restrict__ cl, const float* __restrict__ Wpe,
                 const float* __restrict__ bpe, float* __restrict__ h,
                 unsigned short* __restrict__ hbf) {
  const int p = blockIdx.x, b = blockIdx.y, tid = threadIdx.x;
  __shared__ float patch[16];
  if (tid < 16) patch[tid] = cl[(size_t)b * SEQ + p * 8 + tid];
  __syncthreads();
  for (int d = tid; d < DIM; d += 256) {
    float acc = bpe[d];
#pragma unroll
    for (int j = 0; j < 16; ++j) acc += patch[j] * Wpe[j * DIM + d];
    int i = d >> 1;
    float div = __expf((float)(2 * i) * (-9.210340372f / (float)DIM)); // ln(1e4)
    float ang = (float)p * div;
    acc += (d & 1) ? __cosf(ang) : __sinf(ang);
    size_t off = ((size_t)b * NP + p) * DIM + d;
    h[off]   = acc;
    hbf[off] = f2bf(acc);
  }
}

// ------------------------------------------------------- WMMA bf16 GEMM
// C[M,N] = A[M,K] * B[K,N] (+bias). EPI 0: f32 out.  EPI 1: GELU -> bf16 out.
// Block tile 64x128, 8 waves; each wave owns 16 rows x 64 cols (4 WMMA accs),
// so one A fragment feeds 4 WMMAs. A tile is DMA'd into LDS with the CDNA5
// async path (GLOBAL_LOAD_ASYNC_TO_LDS_B128, ASYNCcnt); B is read coalesced
// and transposed through LDS to n-major for contiguous fragment loads.
template <int EPI>
__global__ __launch_bounds__(256)
void gemm_bf16(const unsigned short* __restrict__ A, const unsigned short* __restrict__ Bw,
               const float* __restrict__ bias, void* __restrict__ Cout,
               int M, int N, int K) {
  __shared__ __align__(16) unsigned short As[64][32];   // [m][k]  4KB
  __shared__ __align__(16) unsigned short Bs[128][32];  // [n][k]  8KB (transposed)
  const int tid  = threadIdx.x;
  const int lane = tid & 31, w = tid >> 5;
  const int wm = w & 3, wn = w >> 2;
  const int m0 = blockIdx.y * 64, n0 = blockIdx.x * 128;
  const int hsel = lane >> 4, lc = lane & 15;
  const int base = hsel * 8;              // CDNA5 16-bit A/B lane K pattern

  // A async-copy coords: each thread moves 16B (one b128) per K-step
  const int ar = tid >> 2;                // 0..63
  const int ac = (tid & 3) * 8;           // 0,8,16,24
  const unsigned a_lds = (unsigned)(size_t)&As[ar][ac];  // 32-bit LDS offset
  // B load coords: each thread reads 16 consecutive n (32B coalesced) for one k
  const int bk = tid >> 3;                // 0..31
  const int bn = (tid & 7) * 16;          // 0..112

  f32x8 acc[4] = {{}, {}, {}, {}};
  for (int k0 = 0; k0 < K; k0 += 32) {
    __syncthreads();
    { // ---- async DMA: A tile global -> LDS (tracked by ASYNCcnt)
      const unsigned short* gp = &A[(size_t)(m0 + ar) * K + k0 + ac];
      asm volatile("global_load_async_to_lds_b128 %0, %1, off"
                   :: "v"(a_lds), "v"(gp) : "memory");
    }
    if (k0 + 32 < K)  // prefetch next B tile (global_prefetch_b8)
      __builtin_prefetch(&Bw[(size_t)(k0 + 32 + bk) * N + n0 + bn], 0, 0);
    { // ---- B tile: coalesced read, transpose-store to n-major LDS
      const unsigned short* gp = &Bw[(size_t)(k0 + bk) * N + n0 + bn];
      u32x4 q0 = *(const u32x4*)gp;
      u32x4 q1 = *(const u32x4*)(gp + 8);
#pragma unroll
      for (int i = 0; i < 4; ++i) {
        unsigned v0 = q0[i], v1 = q1[i];
        Bs[bn + 2 * i    ][bk] = (unsigned short)v0;
        Bs[bn + 2 * i + 1][bk] = (unsigned short)(v0 >> 16);
        Bs[bn + 8 + 2 * i][bk] = (unsigned short)v1;
        Bs[bn + 9 + 2 * i][bk] = (unsigned short)(v1 >> 16);
      }
    }
    asm volatile("s_wait_asynccnt 0" ::: "memory");  // our async A-copy done
    __syncthreads();                                 // publish LDS to all waves
    bf16x16 fa = load_frag(&As[wm * 16 + lc][base], &As[wm * 16 + lc][16 + base]);
#pragma unroll
    for (int t = 0; t < 4; ++t) {
      const int n = wn * 64 + t * 16 + lc;
      bf16x16 fb = load_frag(&Bs[n][base], &Bs[n][16 + base]);
      acc[t] = __builtin_amdgcn_wmma_f32_16x16x32_bf16(false, fa, false, fb,
                                                       (short)0, acc[t], false, false);
    }
  }
#pragma unroll
  for (int t = 0; t < 4; ++t) {
    const int ncol = n0 + wn * 64 + t * 16 + lc;
    const float bv = bias ? bias[ncol] : 0.f;
#pragma unroll
    for (int r = 0; r < 8; ++r) {         // D layout: vgpr r -> row r + 8*hsel
      int m = m0 + wm * 16 + r + 8 * hsel;
      float v = acc[t][r] + bv;
      if (EPI == 1) {
        v = gelu_exact(v);
        ((unsigned short*)Cout)[(size_t)m * N + ncol] = f2bf(v);
      } else {
        ((float*)Cout)[(size_t)m * N + ncol] = v;
      }
    }
  }
}

// ------------------------------------------------------- attention (b,head)
// 4 waves; full K/V for one head live in LDS (CDNA5: 320KB/WGP).
__global__ __launch_bounds__(128)
void attention(const float* __restrict__ qkv, unsigned short* __restrict__ obf) {
  extern __shared__ char smem[];
  unsigned short* Kt = (unsigned short*)smem;             // [256][64]  keys
  unsigned short* Vt = Kt + 256 * 64;                     // [64][256]  V^T
  unsigned short* Qt = Vt + 64 * 256;                     // [4][16][64]
  float*          Sc = (float*)(Qt + 4 * 16 * 64);        // [4][16][256]
  unsigned short* Pr = (unsigned short*)(Sc + 4 * 16 * 256); // [4][16][256]
  float*          Rs = (float*)(Pr + 4 * 16 * 256);       // [4][16] row sums

  const int bh = blockIdx.x;
  const int b = bh >> 3, head = bh & 7;
  const int tid = threadIdx.x;
  const int lane = tid & 31, w = tid >> 5;
  const int hsel = lane >> 4, lc = lane & 15;
  const int base = hsel * 8;
  const size_t rowoff = (size_t)b * NP;

  for (int i = tid; i < 256 * 64; i += 128) {
    int n = i >> 6, d = i & 63;
    float kv = (n < NP) ? qkv[(rowoff + n) * N3D + DIM + head * DHD + d] : 0.f;
    Kt[n * 64 + d] = f2bf(kv);
    int d2 = i >> 8, n2 = i & 255;
    float vv = (n2 < NP) ? qkv[(rowoff + n2) * N3D + 2 * DIM + head * DHD + d2] : 0.f;
    Vt[d2 * 256 + n2] = f2bf(vv);
  }
  __syncthreads();

  unsigned short* Qw = Qt + w * (16 * 64);
  float*          Sw = Sc + w * (16 * 256);
  unsigned short* Pw = Pr + w * (16 * 256);
  float*          Rw = Rs + w * 16;

  for (int qt = w; qt < 16; qt += 4) {
    const int q0 = qt * 16;
    for (int i = lane; i < 16 * 64; i += 32) {     // stage Q tile (bf16)
      int r = i >> 6, d = i & 63;
      int qr = q0 + r;
      float qv = (qr < NP) ? qkv[(rowoff + qr) * N3D + head * DHD + d] : 0.f;
      Qw[r * 64 + d] = f2bf(qv);
    }
    // Q fragments, K-dim 64 => two 32-wide windows
    bf16x16 a0 = load_frag(&Qw[lc * 64 + base],      &Qw[lc * 64 + 16 + base]);
    bf16x16 a1 = load_frag(&Qw[lc * 64 + 32 + base], &Qw[lc * 64 + 48 + base]);
    for (int nt = 0; nt < 16; ++nt) {              // scores S = Q K^T * scale
      const int n = nt * 16 + lc;
      bf16x16 b0 = load_frag(&Kt[n * 64 + base],      &Kt[n * 64 + 16 + base]);
      bf16x16 b1 = load_frag(&Kt[n * 64 + 32 + base], &Kt[n * 64 + 48 + base]);
      f32x8 s = {};
      s = __builtin_amdgcn_wmma_f32_16x16x32_bf16(false, a0, false, b0, (short)0, s, false, false);
      s = __builtin_amdgcn_wmma_f32_16x16x32_bf16(false, a1, false, b1, (short)0, s, false, false);
#pragma unroll
      for (int r = 0; r < 8; ++r)
        Sw[(r + 8 * hsel) * 256 + nt * 16 + lc] = s[r] * 0.125f;  // 1/sqrt(64)
    }
    { // softmax, row = lc, halves split by hsel, combined via shfl_xor(16)
      const float* Srow = Sw + lc * 256;
      const int c0 = hsel * 128;
      float mx = -1e30f;
      for (int c = c0; c < c0 + 128; ++c)
        if (c < NP) mx = fmaxf(mx, Srow[c]);
      mx = fmaxf(mx, __shfl_xor(mx, 16));
      float sum = 0.f;
      unsigned short* Prow = Pw + lc * 256;
      for (int c = c0; c < c0 + 128; ++c) {
        float e = (c < NP) ? __expf(Srow[c] - mx) : 0.f; // mask pad col 255
        sum += e;
        Prow[c] = f2bf(e);                                // unnormalized probs
      }
      sum += __shfl_xor(sum, 16);
      if (lane < 16) Rw[lc] = fmaxf(sum, 1e-20f);
    }
    for (int ct = 0; ct < 4; ++ct) {               // O = P V (normalize after)
      f32x8 acc = {};
#pragma unroll
      for (int kw = 0; kw < 256; kw += 32) {
        bf16x16 pa = load_frag(&Pw[lc * 256 + kw + base], &Pw[lc * 256 + kw + 16 + base]);
        const int nv = ct * 16 + lc;
        bf16x16 pb = load_frag(&Vt[nv * 256 + kw + base], &Vt[nv * 256 + kw + 16 + base]);
        acc = __builtin_amdgcn_wmma_f32_16x16x32_bf16(false, pa, false, pb,
                                                      (short)0, acc, false, false);
      }
#pragma unroll
      for (int r = 0; r < 8; ++r) {
        int m = r + 8 * hsel;
        int qr = q0 + m;
        if (qr < NP)
          obf[(rowoff + qr) * DIM + head * DHD + ct * 16 + lc] = f2bf(acc[r] / Rw[m]);
      }
    }
  }
}

// ----------------------------------------------- residual + bias + LayerNorm
__global__ __launch_bounds__(256)
void resid_ln(const float* __restrict__ hres, const float* __restrict__ delta,
              const float* __restrict__ bias, const float* __restrict__ g,
              const float* __restrict__ bta, float* __restrict__ hout,
              unsigned short* __restrict__ houtbf) {
  const int t = blockIdx.x, tid = threadIdx.x;
  const size_t basei = (size_t)t * DIM;
  const int i0 = tid, i1 = tid + 256;
  float a = hres[basei + i0] + delta[basei + i0] + bias[i0];
  float c = hres[basei + i1] + delta[basei + i1] + bias[i1];
  __shared__ float red[256];
  red[tid] = a + c; __syncthreads();
  for (int off = 128; off > 0; off >>= 1) {
    if (tid < off) red[tid] += red[tid + off];
    __syncthreads();
  }
  float mean = red[0] / (float)DIM;
  __syncthreads();
  float da = a - mean, dc = c - mean;
  red[tid] = da * da + dc * dc; __syncthreads();
  for (int off = 128; off > 0; off >>= 1) {
    if (tid < off) red[tid] += red[tid + off];
    __syncthreads();
  }
  float rstd = rsqrtf(red[0] / (float)DIM + 1e-5f);
  float oa = da * rstd * g[i0] + bta[i0];
  float oc = dc * rstd * g[i1] + bta[i1];
  hout[basei + i0] = oa;  houtbf[basei + i0] = f2bf(oa);
  hout[basei + i1] = oc;  houtbf[basei + i1] = f2bf(oc);
}

// ------------------------------------------------------------- mean pool
__global__ __launch_bounds__(256)
void pool_mean(const float* __restrict__ h, float* __restrict__ pooled) {
  const int b = blockIdx.x, tid = threadIdx.x;
  for (int d = tid; d < DIM; d += 256) {
    float s = 0.f;
    for (int p = 0; p < NP; ++p) s += h[((size_t)b * NP + p) * DIM + d];
    pooled[(size_t)b * DIM + d] = s * (1.f / (float)NP);
  }
}

// -------------------------------------------------------- classification head
__global__ __launch_bounds__(256)
void head_mlp(const float* __restrict__ pooled, const float* __restrict__ Wh1,
              const float* __restrict__ bh1, const float* __restrict__ Wh2,
              const float* __restrict__ bh2, float* __restrict__ out) {
  const int b = blockIdx.x, tid = threadIdx.x;
  __shared__ float hid[256];
  float acc = bh1[tid];
  for (int d = 0; d < DIM; ++d) acc += pooled[(size_t)b * DIM + d] * Wh1[d * 256 + tid];
  hid[tid] = gelu_exact(acc);
  __syncthreads();
  if (tid < 3) {
    float o = bh2[tid];
    for (int j = 0; j < 256; ++j) o += hid[j] * Wh2[j * 3 + tid];
    out[b * 3 + tid] = o;
  }
}

// ----------------------------------------------------------------- launcher
extern "C" void kernel_launch(void* const* d_in, const int* in_sizes, int n_in,
                              void* d_out, int out_size, void* d_ws, size_t ws_size,
                              hipStream_t stream) {
  (void)in_sizes; (void)n_in; (void)out_size; (void)ws_size;
  const float* x    = (const float*)d_in[0];
  const float* Wpe  = (const float*)d_in[1];
  const float* bpe  = (const float*)d_in[2];
  const float* Wqkv = (const float*)d_in[3];
  const float* bqkv = (const float*)d_in[4];
  const float* Wo   = (const float*)d_in[5];
  const float* bo   = (const float*)d_in[6];
  const float* W1   = (const float*)d_in[7];
  const float* b1   = (const float*)d_in[8];
  const float* W2   = (const float*)d_in[9];
  const float* b2   = (const float*)d_in[10];
  const float* ln1g = (const float*)d_in[11];
  const float* ln1b = (const float*)d_in[12];
  const float* ln2g = (const float*)d_in[13];
  const float* ln2b = (const float*)d_in[14];
  const float* Wh1  = (const float*)d_in[15];
  const float* bh1  = (const float*)d_in[16];
  const float* Wh2  = (const float*)d_in[17];
  const float* bh2  = (const float*)d_in[18];
  float* out = (float*)d_out;

  char* ws = (char*)d_ws;
  size_t off = 0;
  auto alloc = [&](size_t bytes) -> char* {
    char* p = ws + off;
    off = (off + bytes + 255) & ~(size_t)255;
    return p;
  };
  float*          cl     = (float*)         alloc((size_t)NB * SEQ * 4);
  float*          h      = (float*)         alloc((size_t)MTOK * DIM * 4);
  unsigned short* hbf    = (unsigned short*)alloc((size_t)MTOK * DIM * 2);
  float*          qkvb   = (float*)         alloc((size_t)MTOK * N3D * 4);
  unsigned short* obf    = (unsigned short*)alloc((size_t)MTOK * DIM * 2);
  float*          c1     = (float*)         alloc((size_t)MTOK * DIM * 4);
  unsigned short* f1bf   = (unsigned short*)alloc((size_t)MTOK * FFD * 2);
  unsigned short* wqkvbf = (unsigned short*)alloc((size_t)NL * DIM * N3D * 2);
  unsigned short* wobf   = (unsigned short*)alloc((size_t)NL * DIM * DIM * 2);
  unsigned short* w1bf   = (unsigned short*)alloc((size_t)NL * DIM * FFD * 2);
  unsigned short* w2bf   = (unsigned short*)alloc((size_t)NL * FFD * DIM * 2);
  float*          pooled = (float*)         alloc((size_t)NB * DIM * 4);

  auto cvt = [&](const float* s, unsigned short* d, size_t n) {
    to_bf16k<<<dim3((unsigned)((n + 255) / 256)), dim3(256), 0, stream>>>(s, d, n);
  };
  cvt(Wqkv, wqkvbf, (size_t)NL * DIM * N3D);
  cvt(Wo,   wobf,   (size_t)NL * DIM * DIM);
  cvt(W1,   w1bf,   (size_t)NL * DIM * FFD);
  cvt(W2,   w2bf,   (size_t)NL * FFD * DIM);

  close_norm<<<dim3(NB), dim3(256), 0, stream>>>(x, cl);
  patch_embed<<<dim3(NP, NB), dim3(256), 0, stream>>>(cl, Wpe, bpe, h, hbf);

  const size_t attn_smem = (size_t)(256 * 64 + 64 * 256 + 4 * 16 * 64) * 2 +
                           (size_t)(4 * 16 * 256) * 4 +
                           (size_t)(4 * 16 * 256) * 2 + (size_t)(4 * 16) * 4; // 172288 B

  for (int l = 0; l < NL; ++l) {
    gemm_bf16<0><<<dim3(N3D / 128, MTOK / 64), dim3(256), 0, stream>>>(
        hbf, wqkvbf + (size_t)l * DIM * N3D, bqkv + (size_t)l * N3D, qkvb, MTOK, N3D, DIM);
    attention<<<dim3(NB * NH), dim3(128), attn_smem, stream>>>(qkvb, obf);
    gemm_bf16<0><<<dim3(DIM / 128, MTOK / 64), dim3(256), 0, stream>>>(
        obf, wobf + (size_t)l * DIM * DIM, nullptr, c1, MTOK, DIM, DIM);
    resid_ln<<<dim3(MTOK), dim3(256), 0, stream>>>(
        h, c1, bo + (size_t)l * DIM, ln1g + (size_t)l * DIM, ln1b + (size_t)l * DIM, h, hbf);
    gemm_bf16<1><<<dim3(FFD / 128, MTOK / 64), dim3(256), 0, stream>>>(
        hbf, w1bf + (size_t)l * DIM * FFD, b1 + (size_t)l * FFD, f1bf, MTOK, FFD, DIM);
    gemm_bf16<0><<<dim3(DIM / 128, MTOK / 64), dim3(256), 0, stream>>>(
        f1bf, w2bf + (size_t)l * FFD * DIM, nullptr, c1, MTOK, DIM, FFD);
    resid_ln<<<dim3(MTOK), dim3(256), 0, stream>>>(
        h, c1, b2 + (size_t)l * DIM, ln2g + (size_t)l * DIM, ln2b + (size_t)l * DIM, h, hbf);
  }

  pool_mean<<<dim3(NB), dim3(256), 0, stream>>>(h, pooled);
  head_mlp<<<dim3(NB), dim3(256), 0, stream>>>(pooled, Wh1, bh1, Wh2, bh2, out);
}
